// Geminal_30391188586536
// MI455X (gfx1250) — matrix operand order repaired and checked by reading, not compiled, not run on hardware
//
#include <hip/hip_runtime.h>
#include <hip/hip_bf16.h>

typedef __attribute__((ext_vector_type(16))) _Float16 v16h;
typedef __attribute__((ext_vector_type(8)))  float    v8f;
typedef __attribute__((ext_vector_type(2)))  float    v2f;

#define PI_F 3.14159265358979323846f
#define L_F  10.0f

// ---------------------------------------------------------------- helpers
__device__ inline float rdist3(const float d[3]) {
    float s0 = __sinf(PI_F * d[0] / L_F);
    float s1 = __sinf(PI_F * d[1] / L_F);
    float s2 = __sinf(PI_F * d[2] / L_F);
    return sqrtf(s0 * s0 + s1 * s1 + s2 * s2) * (L_F / PI_F);
}

// feature f of the 31-dim fourier vector: [r, cos_k(3), sin_k(3)]*5
__device__ inline float feat_of(int f, const float d[3], float rval) {
    if (f == 0) return rval;
    if (f > 30) return 0.f;
    int idx = f - 1;
    int k = idx / 6 + 1;
    int rem = idx % 6;
    int dim = rem % 3;
    float ang = (2.f * PI_F * (float)k / L_F) * d[dim];
    return (rem < 3) ? __cosf(ang) : __sinf(ang);
}

// ------------------------------------------------ d=0 means of raw features
// g2a0[j] = mean_{i<512} ee0[i,j,:], g2b0[j] = mean_{i>=512}, g3_0[i] = mean_j ep0[i,j,:]
__global__ void k_pair_means0(const float* __restrict__ sx,
                              float* __restrict__ g2a0, float* __restrict__ g2b0,
                              float* __restrict__ g3_0) {
    const float* S = sx;
    const float* X = sx + 1024 * 3;
    int t = blockIdx.x;          // 0..1023
    int tid = threadIdx.x;       // 128 threads
    float accA[31], accB[31], accC[31];
    for (int q = 0; q < 31; q++) { accA[q] = 0.f; accB[q] = 0.f; accC[q] = 0.f; }
    float xt0 = X[t * 3], xt1 = X[t * 3 + 1], xt2 = X[t * 3 + 2];
    for (int i = tid; i < 1024; i += blockDim.x) {
        // ee pair (i, t): d = x_i - x_t
        float d[3] = { X[i * 3] - xt0, X[i * 3 + 1] - xt1, X[i * 3 + 2] - xt2 };
        float rv = (i == t) ? 0.f : rdist3(d);
        float* acc = (i < 512) ? accA : accB;
        for (int q = 0; q < 31; q++) acc[q] += feat_of(q, d, rv);
        // ep pair (t, i): d = x_t - s_i
        float dp[3] = { xt0 - S[i * 3], xt1 - S[i * 3 + 1], xt2 - S[i * 3 + 2] };
        float rp = rdist3(dp);
        for (int q = 0; q < 31; q++) accC[q] += feat_of(q, dp, rp);
    }
    __shared__ float red[93];
    for (int q = tid; q < 93; q += blockDim.x) red[q] = 0.f;
    __syncthreads();
    for (int q = 0; q < 31; q++) {
        atomicAdd(&red[q], accA[q]);
        atomicAdd(&red[31 + q], accB[q]);
        atomicAdd(&red[62 + q], accC[q]);
    }
    __syncthreads();
    for (int q = tid; q < 31; q += blockDim.x) {
        g2a0[t * 31 + q] = red[q] * (1.f / 512.f);
        g2b0[t * 31 + q] = red[31 + q] * (1.f / 512.f);
        g3_0[t * 31 + q] = red[62 + q] * (1.f / 1024.f);
    }
}

// ------------------------------------------------ d=0 e-stream MLP (102->64)
__global__ void k_e_mlp0(const float* __restrict__ kp,
                         const float* __restrict__ g2a0, const float* __restrict__ g2b0,
                         const float* __restrict__ g3_0,
                         const float* __restrict__ We0, const float* __restrict__ be0,
                         float* __restrict__ e) {
    int gid = blockIdx.x * blockDim.x + threadIdx.x;   // 1024*64
    int t = gid >> 6, c = gid & 63;
    float acc = be0[c];
    for (int d = 0; d < 3; d++) {
        float kv = kp[d];   // e rows, g1a, g1b are all broadcast kpoints[0]
        acc += kv * (We0[d * 64 + c] + We0[(3 + d) * 64 + c] + We0[(6 + d) * 64 + c]);
    }
    for (int q = 0; q < 31; q++) {
        acc += g2a0[t * 31 + q] * We0[(9 + q) * 64 + c];
        acc += g2b0[t * 31 + q] * We0[(40 + q) * 64 + c];
        acc += g3_0[t * 31 + q] * We0[(71 + q) * 64 + c];
    }
    e[gid] = tanhf(acc);
}

// ------------------------------------------------ pair layer 0: fourier(31) @ W -> tanh -> f16
// one wave per 16-row tile (16 pairs x 16 channels), WMMA f16 K=32 (31 + 1 zero pad)
__global__ void k_pair_layer0(const float* __restrict__ sx,
                              const float* __restrict__ W,     // (31,16)
                              const float* __restrict__ bias,  // (16)
                              _Float16* __restrict__ out,      // 1024*1024*16
                              int is_ee) {
    const float* S = sx;
    const float* X = sx + 1024 * 3;
    int wave = blockIdx.x * (blockDim.x >> 5) + (threadIdx.x >> 5);
    int lane = threadIdx.x & 31;
    int half = lane >> 4, ln = lane & 15;
    long p0 = (long)wave * 16;          // first flattened pair row of tile
    int i = (int)(p0 >> 10);            // constant within tile (1024 % 16 == 0)
    int j = (int)(p0 & 1023) + ln;      // this lane's pair column
    const float* other = is_ee ? X : S;
    float d[3] = { X[i * 3] - other[j * 3],
                   X[i * 3 + 1] - other[j * 3 + 1],
                   X[i * 3 + 2] - other[j * 3 + 2] };
    float rv = (is_ee && i == j) ? 0.f : rdist3(d);
    // A layout (f16 16x32): half0 elems = K{0..7,16..23}, half1 = K{8..15,24..31}
    v16h a, b;
    for (int e = 0; e < 16; e++) {
        int K = (e < 8) ? (8 * half + e) : (16 + 8 * half + (e - 8));
        a[e] = (_Float16)feat_of(K, d, rv);
    }
    // B layout: lane = column N, same K order as A
    for (int e = 0; e < 16; e++) {
        int K = (e < 8) ? (8 * half + e) : (16 + 8 * half + (e - 8));
        b[e] = (_Float16)((K < 31) ? W[K * 16 + ln] : 0.f);
    }
    v8f c = {};
    c = __builtin_amdgcn_wmma_f32_16x16x32_f16(false, a, false, b, (short)0, c, false, false);
    float bv = bias[ln];
    for (int r = 0; r < 8; r++) {
        int M = r + 8 * half;           // D: vgpr r -> rows r / r+8 per lane half
        out[(p0 + M) * 16 + ln] = (_Float16)tanhf(c[r] + bv);
    }
}

// ------------------------------------------------ pair residual layer: ee = tanh(ee@W+b) + ee
// Each wave async-stages its 512B tile (16 rows x 16 ch f16, contiguous) into LDS via
// global_load_async_to_lds_b128 (one 16B chunk per lane), waits on ASYNCcnt, then feeds
// both the WMMA A fragment and the residual add from LDS.
__global__ void k_pair_layer(_Float16* __restrict__ ee,
                             const float* __restrict__ W,     // (16,16)
                             const float* __restrict__ bias) {
    __shared__ _Float16 stage[8][256];   // 8 waves/block, 512B tile each
    int wib = threadIdx.x >> 5;
    int wave = blockIdx.x * (blockDim.x >> 5) + wib;
    int lane = threadIdx.x & 31;
    int half = lane >> 4, ln = lane & 15;
    long p0 = (long)wave * 16;
    // ---- async stage: 32 lanes x 16B = 512B tile
    {
        const _Float16* gbase = ee + p0 * 16;                       // tile base (uniform)
        unsigned lds_off = (unsigned)(unsigned long long)&stage[wib][lane * 8];
        unsigned goff = (unsigned)(lane * 16);                      // byte offset per lane
        asm volatile("global_load_async_to_lds_b128 %0, %1, %2 offset:0"
                     :: "v"(lds_off), "v"(goff), "s"(gbase) : "memory");
        asm volatile("s_wait_asynccnt 0x0" ::: "memory");
    }
    const _Float16* tile = &stage[wib][0];
    v16h a, b;
    for (int e = 0; e < 16; e++) {
        int K = (e < 8) ? (8 * half + e) : (16 + 8 * half + (e - 8));
        a[e] = (K < 16) ? tile[ln * 16 + K] : (_Float16)0.f;
    }
    for (int e = 0; e < 16; e++) {
        int K = (e < 8) ? (8 * half + e) : (16 + 8 * half + (e - 8));
        b[e] = (_Float16)((K < 16) ? W[K * 16 + ln] : 0.f);
    }
    v8f c = {};
    c = __builtin_amdgcn_wmma_f32_16x16x32_f16(false, a, false, b, (short)0, c, false, false);
    float bv = bias[ln];
    for (int r = 0; r < 8; r++) {
        int M = r + 8 * half;
        float old = (float)tile[M * 16 + ln];
        ee[(p0 + M) * 16 + ln] = (_Float16)(tanhf(c[r] + bv) + old);
    }
}

// ------------------------------------------------ column means of ee (over i halves)
__global__ void k_colmeans16(const _Float16* __restrict__ ee,
                             float* __restrict__ g2a, float* __restrict__ g2b) {
    int jb = blockIdx.x;                 // 64 blocks of 16 j's
    int jl = threadIdx.x >> 4;
    int cc = threadIdx.x & 15;
    int j = jb * 16 + jl;
    float aa = 0.f, bb = 0.f;
    for (int i = 0; i < 512; i++) {
        __builtin_prefetch(&ee[((long)(i + 8) * 1024 + j) * 16 + cc], 0, 1);
        aa += (float)ee[((long)i * 1024 + j) * 16 + cc];
    }
    for (int i = 512; i < 1024; i++) {
        bb += (float)ee[((long)i * 1024 + j) * 16 + cc];
    }
    g2a[j * 16 + cc] = aa * (1.f / 512.f);
    g2b[j * 16 + cc] = bb * (1.f / 512.f);
}

// ------------------------------------------------ row means of ep (over j)
__global__ void k_rowmeans16(const _Float16* __restrict__ ep, float* __restrict__ g3) {
    int i = blockIdx.x;
    int slot = threadIdx.x >> 4, cc = threadIdx.x & 15;
    float acc = 0.f;
    for (int j = slot; j < 1024; j += 16)
        acc += (float)ep[((long)i * 1024 + j) * 16 + cc];
    __shared__ float red[256];
    red[threadIdx.x] = acc;
    __syncthreads();
    if (slot == 0) {
        float s = 0.f;
        for (int q = 0; q < 16; q++) s += red[q * 16 + cc];
        g3[i * 16 + cc] = s * (1.f / 1024.f);
    }
}

// ------------------------------------------------ e half-means (g1a | g1b)
__global__ void k_e_means(const float* __restrict__ e, float* __restrict__ g1) {
    int base = blockIdx.x * 512;         // block 0 -> g1a, block 1 -> g1b
    int c = threadIdx.x & 63, slot = threadIdx.x >> 6;
    float acc = 0.f;
    for (int t = slot; t < 512; t += 4) acc += e[(base + t) * 64 + c];
    __shared__ float red[256];
    red[threadIdx.x] = acc;
    __syncthreads();
    if (slot == 0) {
        float s = red[c] + red[64 + c] + red[128 + c] + red[192 + c];
        g1[blockIdx.x * 64 + c] = s * (1.f / 512.f);
    }
}

// ------------------------------------------------ e-stream layer (240 -> 64, residual)
__global__ void k_e_layer(const float* __restrict__ e_in, const float* __restrict__ g1,
                          const float* __restrict__ g2a, const float* __restrict__ g2b,
                          const float* __restrict__ g3,
                          const float* __restrict__ W, const float* __restrict__ bvec,
                          float* __restrict__ e_out) {
    int gid = blockIdx.x * blockDim.x + threadIdx.x;   // 1024*64
    int t = gid >> 6, c = gid & 63;
    float acc = bvec[c];
    for (int q = 0; q < 64; q++) acc += e_in[t * 64 + q] * W[q * 64 + c];
    for (int q = 0; q < 64; q++) acc += g1[q] * W[(64 + q) * 64 + c];
    for (int q = 0; q < 64; q++) acc += g1[64 + q] * W[(128 + q) * 64 + c];
    for (int q = 0; q < 16; q++) acc += g2a[t * 16 + q] * W[(192 + q) * 64 + c];
    for (int q = 0; q < 16; q++) acc += g2b[t * 16 + q] * W[(208 + q) * 64 + c];
    for (int q = 0; q < 16; q++) acc += g3[t * 16 + q] * W[(224 + q) * 64 + c];
    e_out[gid] = tanhf(acc) + e_in[gid];
}

// ------------------------------------------------ orbitals (complex 1024x64)
__global__ void k_orb(const float* __restrict__ e,
                      const float* __restrict__ Wre, const float* __restrict__ Wim,
                      const float* __restrict__ bre, const float* __restrict__ bim,
                      float* __restrict__ orb_re, float* __restrict__ orb_im) {
    int gid = blockIdx.x * blockDim.x + threadIdx.x;
    int t = gid >> 6, c = gid & 63;
    float ar = bre[c], ai = bim[c];
    for (int q = 0; q < 64; q++) {
        float ev = e[t * 64 + q];
        ar += ev * Wre[q * 64 + c];
        ai += ev * Wim[q * 64 + c];
    }
    orb_re[gid] = ar;
    orb_im[gid] = ai;
}

// ------------------------------------------------ T[k,i,b] = sum_a orb_up[i,a] * wdet[k,a,b]  (wdet real)
__global__ void k_T(const float* __restrict__ orb_re, const float* __restrict__ orb_im,
                    const float* __restrict__ wdet,
                    float* __restrict__ T_re, float* __restrict__ T_im) {
    int gid = blockIdx.x * blockDim.x + threadIdx.x;   // 4*512*64
    int b = gid & 63;
    int i = (gid >> 6) & 511;
    int k = gid >> 15;
    const float* wr = wdet + (long)k * 64 * 64;
    float ar = 0.f, ai = 0.f;
    for (int a = 0; a < 64; a++) {
        float w = wr[a * 64 + b];
        ar += orb_re[i * 64 + a] * w;
        ai += orb_im[i * 64 + a] * w;
    }
    T_re[gid] = ar;
    T_im[gid] = ai;
}

// ------------------------------------------------ phi[k,i,j] = T[k,i,:] . orb_dn[j,:] + 1
__global__ void k_phi(const float* __restrict__ T_re, const float* __restrict__ T_im,
                      const float* __restrict__ orb_re, const float* __restrict__ orb_im,
                      float2* __restrict__ phi) {
    int gid = blockIdx.x * blockDim.x + threadIdx.x;   // 4*512*512
    int j = gid & 511;
    int i = (gid >> 9) & 511;
    int k = gid >> 18;
    const float* tr = T_re + ((long)k * 512 + i) * 64;
    const float* ti = T_im + ((long)k * 512 + i) * 64;
    const float* dr = orb_re + (512 + j) * 64;
    const float* di = orb_im + (512 + j) * 64;
    float pr = 1.f, pim = 0.f;
    for (int b = 0; b < 64; b++) {
        float a = tr[b], bb = ti[b], c = dr[b], d = di[b];
        pr += a * c - bb * d;
        pim += a * d + bb * c;
    }
    phi[gid] = make_float2(pr, pim);
}

// ------------------------------------------------ z = e @ bf_w + x
__global__ void k_z(const float* __restrict__ e, const float* __restrict__ sx,
                    const float* __restrict__ bf_w, float* __restrict__ z) {
    int t = blockIdx.x * blockDim.x + threadIdx.x;
    if (t >= 1024) return;
    const float* X = sx + 1024 * 3;
    for (int d = 0; d < 3; d++) {
        float acc = X[t * 3 + d];
        for (int q = 0; q < 64; q++) acc += e[t * 64 + q] * bf_w[q * 3 + d];
        z[t * 3 + d] = acc;
    }
}

// ------------------------------------------------ fdet (tiny MLP, one block)
__global__ void k_fdet(const float* __restrict__ kp,
                       const float* __restrict__ w1, const float* __restrict__ b1,
                       const float* __restrict__ w2, const float* __restrict__ b2,
                       float* __restrict__ fdet) {
    __shared__ float h[64];
    int tid = threadIdx.x;
    if (tid < 64) {
        float acc = b1[tid];
        for (int d = 0; d < 3; d++) acc += kp[d] * w1[d * 64 + tid];
        h[tid] = tanhf(acc);
    }
    __syncthreads();
    for (int idx = tid; idx < 4 * 512; idx += blockDim.x) {
        int k = idx >> 9, a = idx & 511;
        float v;
        if (a == 0) {
            v = 1.f;
        } else {
            int col = k * 511 + (a - 1);
            float acc = b2[col];
            for (int q = 0; q < 64; q++) acc += h[q] * w2[q * 2044 + col];
            v = (acc > 20.f) ? acc : logf(1.f + expf(acc));   // softplus
        }
        fdet[idx] = v;
    }
}

// ------------------------------------------------ plane-wave phase matrices
// dup[i][a] = norm*exp(i kp_a . z_i)   (i<512, a<512)
// ddn[j][a] = conj(norm*exp(i kp_{512+a} . z_{512+j}))  -- conj baked in
__global__ void k_Dphases(const float* __restrict__ kp, const float* __restrict__ z,
                          float* __restrict__ dup_re, float* __restrict__ dup_im,
                          float* __restrict__ ddn_re, float* __restrict__ ddn_im) {
    int gid = blockIdx.x * blockDim.x + threadIdx.x;   // 512*512
    int a = gid & 511, i = gid >> 9;
    const float norm = 0.0316227766016838f;            // L^{-3/2}
    {
        float th = kp[a * 3] * z[i * 3] + kp[a * 3 + 1] * z[i * 3 + 1] + kp[a * 3 + 2] * z[i * 3 + 2];
        dup_re[(long)i * 512 + a] = norm * __cosf(th);
        dup_im[(long)i * 512 + a] = norm * __sinf(th);
    }
    {
        const float* kd = kp + (512 + a) * 3;
        const float* zj = z + (512 + i) * 3;
        float th = kd[0] * zj[0] + kd[1] * zj[1] + kd[2] * zj[2];
        ddn_re[(long)i * 512 + a] = norm * __cosf(th);
        ddn_im[(long)i * 512 + a] = -norm * __sinf(th);
    }
}

// ------------------------------------------------ M[k] = (fdet-scaled D_up^T . conj(D_dn)) * phi[k]
// complex GEMM via 4 real f32 WMMA chains (16x16x4), one wave per 16x16 output tile.
// f32 WMMA does not support A/B operand negation (NEG bits must be 0), so the
// -Ai*Bi term uses a VALU-negated copy of the Bi fragment.
__global__ void k_Mtile(const float* __restrict__ dup_re, const float* __restrict__ dup_im,
                        const float* __restrict__ ddn_re, const float* __restrict__ ddn_im,
                        const float* __restrict__ fdet, const float2* __restrict__ phi,
                        float2* __restrict__ Mout) {
    int wave = blockIdx.x * (blockDim.x >> 5) + (threadIdx.x >> 5);  // 4096 waves
    int lane = threadIdx.x & 31;
    int half = lane >> 4, ln = lane & 15;
    int k = wave >> 10;
    int ti = (wave >> 5) & 31, tj = wave & 31;
    int i = ti * 16 + ln;   // A row for this lane
    int j = tj * 16 + ln;   // B column for this lane
    const float* fd = fdet + k * 512;
    v8f cRe = {}, cIm = {};
    for (int a0 = 0; a0 < 512; a0 += 4) {
        int ks = a0 + 2 * half;                 // f32 A/B layout: K = 2*half + elem
        float f0 = fd[ks], f1 = fd[ks + 1];
        v2f ar = { f0 * dup_re[(long)i * 512 + ks], f1 * dup_re[(long)i * 512 + ks + 1] };
        v2f ai = { f0 * dup_im[(long)i * 512 + ks], f1 * dup_im[(long)i * 512 + ks + 1] };
        v2f br = { ddn_re[(long)j * 512 + ks], ddn_re[(long)j * 512 + ks + 1] };
        v2f bi = { ddn_im[(long)j * 512 + ks], ddn_im[(long)j * 512 + ks + 1] };
        v2f nbi = { -bi.x, -bi.y };
        cRe = __builtin_amdgcn_wmma_f32_16x16x4_f32(false, ar, false, br,  (short)0, cRe, false, false);
        cRe = __builtin_amdgcn_wmma_f32_16x16x4_f32(false, ai, false, nbi, (short)0, cRe, false, false);
        cIm = __builtin_amdgcn_wmma_f32_16x16x4_f32(false, ar, false, bi,  (short)0, cIm, false, false);
        cIm = __builtin_amdgcn_wmma_f32_16x16x4_f32(false, ai, false, br,  (short)0, cIm, false, false);
    }
    for (int r = 0; r < 8; r++) {
        int M = r + 8 * half;
        long oi = ((long)k * 512 + (ti * 16 + M)) * 512 + (tj * 16 + ln);
        float2 ph = phi[oi];
        float dre = cRe[r], dim = cIm[r];
        Mout[oi] = make_float2(dre * ph.x - dim * ph.y, dre * ph.y + dim * ph.x);
    }
}

// ------------------------------------------------ complex LU slogdet, one block per k
__global__ void k_lu(float2* __restrict__ Mat, float* __restrict__ logabs,
                     float2* __restrict__ phase) {
    int k = blockIdx.x;
    float2* A = Mat + (long)k * 512 * 512;
    int tid = threadIdx.x;   // 256
    __shared__ float  s_red[256];
    __shared__ int    s_idx[256];
    __shared__ float2 s_piv_row[512];
    __shared__ float2 s_d;
    __shared__ float  s_logabs;
    __shared__ float2 s_phase;
    if (tid == 0) { s_logabs = 0.f; s_phase = make_float2(1.f, 0.f); }
    __syncthreads();
    for (int col = 0; col < 512; col++) {
        // partial-pivot search over rows >= col
        float best = -1.f; int bidx = col;
        for (int r = col + tid; r < 512; r += 256) {
            float2 v = A[(long)r * 512 + col];
            float m = v.x * v.x + v.y * v.y;
            if (m > best) { best = m; bidx = r; }
        }
        s_red[tid] = best; s_idx[tid] = bidx;
        __syncthreads();
        for (int s = 128; s > 0; s >>= 1) {
            if (tid < s && s_red[tid + s] > s_red[tid]) {
                s_red[tid] = s_red[tid + s];
                s_idx[tid] = s_idx[tid + s];
            }
            __syncthreads();
        }
        int piv = s_idx[0];
        if (piv != col) {
            for (int c2 = tid; c2 < 512; c2 += 256) {
                float2 t1 = A[(long)col * 512 + c2];
                A[(long)col * 512 + c2] = A[(long)piv * 512 + c2];
                A[(long)piv * 512 + c2] = t1;
            }
        }
        __syncthreads();
        if (tid == 0) {
            float2 d = A[(long)col * 512 + col];
            float m2 = d.x * d.x + d.y * d.y;
            float im = rsqrtf(m2);
            s_logabs += 0.5f * logf(m2);
            float pr = d.x * im, pim = d.y * im;
            float2 ph = s_phase;
            float2 np = make_float2(ph.x * pr - ph.y * pim, ph.x * pim + ph.y * pr);
            if (piv != col) { np.x = -np.x; np.y = -np.y; }
            s_phase = np;
            s_d = d;
        }
        __syncthreads();
        float2 d = s_d;
        float m2 = d.x * d.x + d.y * d.y;
        float inv_r = d.x / m2, inv_i = -d.y / m2;
        for (int c2 = col + tid; c2 < 512; c2 += 256) s_piv_row[c2] = A[(long)col * 512 + c2];
        __syncthreads();
        for (int r = col + 1 + tid; r < 512; r += 256) {
            float2 lv = A[(long)r * 512 + col];
            float fr = lv.x * inv_r - lv.y * inv_i;
            float fi = lv.x * inv_i + lv.y * inv_r;
            for (int c2 = col + 1; c2 < 512; c2++) {
                float2 u = s_piv_row[c2];
                float2 w = A[(long)r * 512 + c2];
                w.x -= fr * u.x - fi * u.y;
                w.y -= fr * u.y + fi * u.x;
                A[(long)r * 512 + c2] = w;
            }
        }
        __syncthreads();
    }
    if (tid == 0) { logabs[k] = s_logabs; phase[k] = s_phase; }
}

// ------------------------------------------------ final log-sum of determinants
__global__ void k_final(const float* __restrict__ logabs, const float2* __restrict__ phase,
                        float* __restrict__ out, int out_size) {
    if (threadIdx.x != 0 || blockIdx.x != 0) return;
    float maxl = logabs[0];
    for (int k = 1; k < 4; k++) maxl = fmaxf(maxl, logabs[k]);
    float dr = 0.f, di = 0.f;
    for (int k = 0; k < 4; k++) {
        float s = expf(logabs[k] - maxl);
        dr += phase[k].x * s;
        di += phase[k].y * s;
    }
    float mag = sqrtf(dr * dr + di * di);
    out[0] = logf(mag) + maxl;
    if (out_size > 1) out[1] = atan2f(di, dr);
}

// ================================================================ host
extern "C" void kernel_launch(void* const* d_in, const int* in_sizes, int n_in,
                              void* d_out, int out_size, void* d_ws, size_t ws_size,
                              hipStream_t stream) {
    (void)in_sizes; (void)n_in; (void)ws_size;
    const float* sx       = (const float*)d_in[0];
    const float* kp       = (const float*)d_in[1];
    const float* we0      = (const float*)d_in[2];
    const float* be0      = (const float*)d_in[3];
    const float* we_rest  = (const float*)d_in[4];   // 3 x 240 x 64
    const float* be_rest  = (const float*)d_in[5];   // 3 x 64
    const float* wee0     = (const float*)d_in[6];
    const float* bee0     = (const float*)d_in[7];
    const float* wee_rest = (const float*)d_in[8];   // 2 x 16 x 16
    const float* bee_rest = (const float*)d_in[9];
    const float* wep0     = (const float*)d_in[10];
    const float* bep0     = (const float*)d_in[11];
    const float* wep_rest = (const float*)d_in[12];
    const float* bep_rest = (const float*)d_in[13];
    const float* orb_w_re = (const float*)d_in[14];
    const float* orb_w_im = (const float*)d_in[15];
    const float* orb_b_re = (const float*)d_in[16];
    const float* orb_b_im = (const float*)d_in[17];
    const float* w_det    = (const float*)d_in[18];
    const float* bf_w     = (const float*)d_in[19];
    const float* mlp_w1   = (const float*)d_in[20];
    const float* mlp_b1   = (const float*)d_in[21];
    const float* mlp_w2   = (const float*)d_in[22];
    const float* mlp_b2   = (const float*)d_in[23];

    char* ws = (char*)d_ws;
    size_t off = 0;
    auto carve = [&](size_t n) -> char* {
        char* p = ws + off;
        off = (off + n + 255) & ~(size_t)255;
        return p;
    };
    _Float16* ee    = (_Float16*)carve(1024ull * 1024 * 16 * 2);
    _Float16* ep    = (_Float16*)carve(1024ull * 1024 * 16 * 2);
    float* g2a0     = (float*)carve(1024 * 31 * 4);
    float* g2b0     = (float*)carve(1024 * 31 * 4);
    float* g3_0     = (float*)carve(1024 * 31 * 4);
    float* g2a      = (float*)carve(1024 * 16 * 4);
    float* g2b      = (float*)carve(1024 * 16 * 4);
    float* g3       = (float*)carve(1024 * 16 * 4);
    float* eA       = (float*)carve(1024 * 64 * 4);
    float* eB       = (float*)carve(1024 * 64 * 4);
    float* g1       = (float*)carve(128 * 4);
    float* orb_re   = (float*)carve(1024 * 64 * 4);
    float* orb_im   = (float*)carve(1024 * 64 * 4);
    float* T_re     = (float*)carve(4 * 512 * 64 * 4);
    float* T_im     = (float*)carve(4 * 512 * 64 * 4);
    float2* phi     = (float2*)carve(4ull * 512 * 512 * 8);
    float* zbuf     = (float*)carve(1024 * 3 * 4);
    float* fdet     = (float*)carve(4 * 512 * 4);
    float* dup_re   = (float*)carve(512ull * 512 * 4);
    float* dup_im   = (float*)carve(512ull * 512 * 4);
    float* ddn_re   = (float*)carve(512ull * 512 * 4);
    float* ddn_im   = (float*)carve(512ull * 512 * 4);
    float2* Mmat    = (float2*)carve(4ull * 512 * 512 * 8);
    float* logabsW  = (float*)carve(4 * 4);
    float2* phaseW  = (float2*)carve(4 * 8);

    // ---- d=0
    k_pair_means0<<<1024, 128, 0, stream>>>(sx, g2a0, g2b0, g3_0);
    k_e_mlp0<<<256, 256, 0, stream>>>(kp, g2a0, g2b0, g3_0, we0, be0, eA);
    k_pair_layer0<<<8192, 256, 0, stream>>>(sx, wee0, bee0, ee, 1);
    k_pair_layer0<<<8192, 256, 0, stream>>>(sx, wep0, bep0, ep, 0);

    // ---- d=1
    k_colmeans16<<<64, 256, 0, stream>>>(ee, g2a, g2b);
    k_rowmeans16<<<1024, 256, 0, stream>>>(ep, g3);
    k_e_means<<<2, 256, 0, stream>>>(eA, g1);
    k_e_layer<<<256, 256, 0, stream>>>(eA, g1, g2a, g2b, g3, we_rest, be_rest, eB);
    k_pair_layer<<<8192, 256, 0, stream>>>(ee, wee_rest, bee_rest);
    k_pair_layer<<<8192, 256, 0, stream>>>(ep, wep_rest, bep_rest);

    // ---- d=2
    k_colmeans16<<<64, 256, 0, stream>>>(ee, g2a, g2b);
    k_rowmeans16<<<1024, 256, 0, stream>>>(ep, g3);
    k_e_means<<<2, 256, 0, stream>>>(eB, g1);
    k_e_layer<<<256, 256, 0, stream>>>(eB, g1, g2a, g2b, g3, we_rest + 240 * 64, be_rest + 64, eA);
    k_pair_layer<<<8192, 256, 0, stream>>>(ee, wee_rest + 256, bee_rest + 16);
    k_pair_layer<<<8192, 256, 0, stream>>>(ep, wep_rest + 256, bep_rest + 16);

    // ---- final e layer
    k_colmeans16<<<64, 256, 0, stream>>>(ee, g2a, g2b);
    k_rowmeans16<<<1024, 256, 0, stream>>>(ep, g3);
    k_e_means<<<2, 256, 0, stream>>>(eA, g1);
    k_e_layer<<<256, 256, 0, stream>>>(eA, g1, g2a, g2b, g3, we_rest + 2 * 240 * 64, be_rest + 128, eB);

    // ---- orbitals / phi / determinants
    k_orb<<<256, 256, 0, stream>>>(eB, orb_w_re, orb_w_im, orb_b_re, orb_b_im, orb_re, orb_im);
    k_T<<<512, 256, 0, stream>>>(orb_re, orb_im, w_det, T_re, T_im);
    k_phi<<<4096, 256, 0, stream>>>(T_re, T_im, orb_re, orb_im, phi);
    k_z<<<4, 256, 0, stream>>>(eB, sx, bf_w, zbuf);
    k_fdet<<<1, 256, 0, stream>>>(kp, mlp_w1, mlp_b1, mlp_w2, mlp_b2, fdet);
    k_Dphases<<<1024, 256, 0, stream>>>(kp, zbuf, dup_re, dup_im, ddn_re, ddn_im);
    k_Mtile<<<512, 256, 0, stream>>>(dup_re, dup_im, ddn_re, ddn_im, fdet, phi, Mmat);
    k_lu<<<4, 256, 0, stream>>>(Mmat, logabsW, phaseW);
    k_final<<<1, 1, 0, stream>>>(logabsW, phaseW, (float*)d_out, out_size);
}
